// CgpHmmCell_13529146982863
// MI455X (gfx1250) — matrix operand order repaired and checked by reading, not compiled, not run on hardware
//
#include <hip/hip_runtime.h>
#include <hip/hip_bf16.h>
#include <math.h>

// ---------------- problem constants (match reference) ----------------
#define SDIM   612          // number of HMM states
#define SPAD   616          // padded row stride for Bt / bands
#define ECOLS  101          // E+1 emission columns
#define NBATCH 32
#define TLEN   8192
#define NTHR   640          // 20 wave32s, one thread per state
#define NWAVE  (NTHR / 32)
#define NSC    (611 * 25)   // (S-1) states x 25 contexts for emission softmax

typedef __attribute__((ext_vector_type(16))) _Float16 v16h;
typedef __attribute__((ext_vector_type(8)))  float    v8f;

// ---------------- Tensor Data Mover availability ---------------------
#if defined(__has_builtin)
#  if __has_builtin(__builtin_amdgcn_tensor_load_to_lds)
#    define HAVE_TDM 1
#  endif
#endif

#ifdef HAVE_TDM
typedef unsigned int u32x4 __attribute__((ext_vector_type(4)));
typedef int          i32x4 __attribute__((ext_vector_type(4)));
typedef int          i32x8 __attribute__((ext_vector_type(8)));

// One wave issues a single TDM 2D tile load: 101 rows x 616 dwords,
// tensor_dim0_stride = 616 -> densely packed into LDS (same linear layout
// as the source).  D# packing per CDNA5 ISA section 8.3/8.4.
__device__ __forceinline__ void tdm_load_bt(const float* gsrc, float* ldst) {
    unsigned long long ga = (unsigned long long)(uintptr_t)gsrc;
    unsigned int ldsOff   = (unsigned int)(uintptr_t)ldst;  // addr[31:0] = LDS offset
    u32x4 g0;
    g0.x = 1u;                                              // count=1 (valid), user mode
    g0.y = ldsOff;                                          // lds_addr
    g0.z = (unsigned int)(ga & 0xFFFFFFFFu);                // global_addr[31:0]
    g0.w = (unsigned int)((ga >> 32) & 0x01FFFFFFu)         // global_addr[56:32]
         | (2u << 30);                                      // type = 2 ("image")
    i32x8 g1;
    g1[0] = 0x00020000;                 // workgroup_mask=0, data_size=2 (4 bytes)
    g1[1] = (SPAD  & 0xFFFF) << 16;     // tensor_dim0[15:0]   (abar addr = 0)
    g1[2] = (ECOLS & 0xFFFF) << 16;     // tensor_dim0[31:16]=0 | tensor_dim1[15:0]
    g1[3] = (SPAD << 16);               // tensor_dim1[31:16]=0 | tile_dim0 = 616
    g1[4] = ECOLS;                      // tile_dim1 = 101, tile_dim2 = 0
    g1[5] = SPAD;                       // tensor_dim0_stride[31:0]
    g1[6] = 0;                          // stride0 hi, stride1 lo (unused, 2D)
    g1[7] = 0;
    i32x4 z4 = {0, 0, 0, 0};
#if __clang_major__ >= 23
    i32x8 z8 = {0, 0, 0, 0, 0, 0, 0, 0};
    __builtin_amdgcn_tensor_load_to_lds(g0, g1, z4, z4, z8, 0);
#else
    __builtin_amdgcn_tensor_load_to_lds(g0, g1, z4, z4, 0);
#endif
}

__device__ __forceinline__ void tdm_wait0() {
#if __has_builtin(__builtin_amdgcn_s_wait_tensorcnt)
    __builtin_amdgcn_s_wait_tensorcnt(0);
#else
    asm volatile("s_wait_tensorcnt 0x0" ::: "memory");
#endif
}
#endif  // HAVE_TDM

// ---------------------------------------------------------------------
// Kernel 0: zero-fill Bt region (terminal column/row stay 0)
// ---------------------------------------------------------------------
__global__ void zero_f32(float* __restrict__ p, int n) {
    int i = blockIdx.x * blockDim.x + threadIdx.x;
    if (i < n) p[i] = 0.0f;
}

// ---------------------------------------------------------------------
// Kernel 1: transition bands.  Row i of A has softmax over logits
// {tk[3i],tk[3i+1],tk[3i+2],1.0} at cols i..i+3 (mod S).
// band[k][(i+k)%S] = p_k  so that (a@A)[c] = sum_k band[k][c]*a[(c-k)%S].
// ---------------------------------------------------------------------
__global__ void build_bands(const float* __restrict__ tk, float* __restrict__ bands) {
    int i = blockIdx.x * blockDim.x + threadIdx.x;
    if (i >= SDIM) return;
    float l0 = tk[3 * i], l1 = tk[3 * i + 1], l2 = tk[3 * i + 2], l3 = 1.0f;
    float m  = fmaxf(fmaxf(l0, l1), fmaxf(l2, l3));
    float e0 = expf(l0 - m), e1 = expf(l1 - m), e2 = expf(l2 - m), e3 = expf(l3 - m);
    float inv = 1.0f / (e0 + e1 + e2 + e3);
    bands[0 * SPAD + i]              = e0 * inv;
    bands[1 * SPAD + (i + 1) % SDIM] = e1 * inv;
    bands[2 * SPAD + (i + 2) % SDIM] = e2 * inv;
    bands[3 * SPAD + (i + 3) % SDIM] = e3 * inv;
}

// ---------------------------------------------------------------------
// Kernel 2: emission matrix Bt [ECOLS][SPAD] (= B^T).  Grouped softmax
// over 4 letters per (state s<611, context c).  c==0 -> const logits 1.0,
// c>=1 -> ek[(s*24 + c-1)*4 + l].  Terminal: Bt[100][611] = 1.
// ---------------------------------------------------------------------
__global__ void build_emis(const float* __restrict__ ek, float* __restrict__ Bt) {
    int idx = blockIdx.x * blockDim.x + threadIdx.x;
    if (idx < NSC) {
        int s = idx / 25, c = idx % 25;
        float l0, l1, l2, l3;
        if (c == 0) {
            l0 = l1 = l2 = l3 = 1.0f;
        } else {
            const float* p = ek + ((s * 24) + (c - 1)) * 4;
            l0 = p[0]; l1 = p[1]; l2 = p[2]; l3 = p[3];
        }
        float m  = fmaxf(fmaxf(l0, l1), fmaxf(l2, l3));
        float e0 = expf(l0 - m), e1 = expf(l1 - m), e2 = expf(l2 - m), e3 = expf(l3 - m);
        float inv = 1.0f / (e0 + e1 + e2 + e3);
        int e = c * 4;
        Bt[(e + 0) * SPAD + s] = e0 * inv;
        Bt[(e + 1) * SPAD + s] = e1 * inv;
        Bt[(e + 2) * SPAD + s] = e2 * inv;
        Bt[(e + 3) * SPAD + s] = e3 * inv;
    } else if (idx == NSC) {
        Bt[100 * SPAD + 611] = 1.0f;   // terminal state emits terminal symbol
    }
}

// ---------------------------------------------------------------------
// Kernel 3: recover obs[b,t] = onehot_row . iota via f16 WMMA.
// Per wave: one 16-row tile, K padded 101->128, 4x v_wmma_f32_16x16x32_f16.
// One-hot values {0,1} and weights <128 are exact in f16 -> exact result.
// Fragment layouts per CDNA5 ISA 7.12.2 (16-bit A 16x32, B 32x16, f32 C).
// ---------------------------------------------------------------------
#define TK 128
__global__ __launch_bounds__(256) void obs_extract_wmma(const float* __restrict__ x,
                                                        int* __restrict__ obs) {
    __shared__ _Float16 sm[8][16 * TK];            // 32 KB: per-wave staging tile
    const int lane = threadIdx.x & 31;
    const int wv   = threadIdx.x >> 5;
    const int tile = blockIdx.x * 8 + wv;          // exact cover: B*T/16 tiles
    _Float16* st = sm[wv];
    const long rowBase = (long)tile * 16;

    // stage 16 rows x 101 f32 -> f16, zero-pad K to 128 (coalesced per row)
    for (int rr = 0; rr < 16; ++rr) {
        const float* src = x + (rowBase + rr) * ECOLS;
        for (int k = lane; k < TK; k += 32)
            st[rr * TK + k] = (k < ECOLS) ? (_Float16)src[k] : (_Float16)0.0f;
    }
    __syncthreads();   // uniform across all 8 waves

    const int laneHi = (lane >= 16) ? 8 : 0;
    const int m      = lane & 15;

    // base weight fragment (chunk kb adds a scalar splat): B 32x16 f16 layout,
    // same K order as A: K = j (+8 if j>=8), +8 more for lanes 16-31.
    v16h bbase;
#pragma unroll
    for (int j = 0; j < 16; ++j)
        bbase[j] = (_Float16)(j + ((j >= 8) ? 8 : 0) + laneHi);

    v8f c = {};
    for (int kb = 0; kb < TK; kb += 32) {
        v16h a;
#pragma unroll
        for (int j = 0; j < 16; ++j) {
            // A 16x32 f16 layout: lane<16: K = j (+8 if j>=8); lane>=16: +8 more
            int kl = j + ((j >= 8) ? 8 : 0) + laneHi;
            a[j] = st[m * TK + kb + kl];
        }
        v16h b = bbase + (_Float16)kb;             // weights for this K-chunk
        c = __builtin_amdgcn_wmma_f32_16x16x32_f16(false, a, false, b,
                                                   (short)0, c, false, false);
    }
    // C layout: VGPR r -> (M = r + (lane<16?0:8), N = lane&15); take column N=0
    if (lane == 0 || lane == 16) {
        int mh = (lane == 16) ? 8 : 0;
#pragma unroll
        for (int r = 0; r < 8; ++r)
            obs[rowBase + mh + r] = (int)(c[r] + 0.5f);
    }
}

// ---------------------------------------------------------------------
// block-wide sum over 640 threads (20 wave32s)
// ---------------------------------------------------------------------
__device__ __forceinline__ float block_sum(float v, float* red) {
    const int tid = threadIdx.x;
#pragma unroll
    for (int off = 16; off > 0; off >>= 1) v += __shfl_xor(v, off, 32);
    if ((tid & 31) == 0) red[tid >> 5] = v;
    __syncthreads();
    if (tid < 32) {
        float p = (tid < NWAVE) ? red[tid] : 0.0f;
#pragma unroll
        for (int off = 16; off > 0; off >>= 1) p += __shfl_xor(p, off, 32);
        if (tid == 0) red[32] = p;
    }
    __syncthreads();
    return red[32];
}

// ---------------------------------------------------------------------
// Kernel 4: persistent forward recurrence.  One workgroup per sequence;
// Bt (249KB, DMA'd in by the TDM) + obs stream (32KB) + double-buffered
// alpha live in LDS (~287KB of the WGP's 320KB).  One thread per state,
// banded 4-tap update, one barrier per step, renormalize every 8 steps.
// ---------------------------------------------------------------------
__global__ __launch_bounds__(NTHR) void forward_kernel(
        const float* __restrict__ Btg, const float* __restrict__ bandsg,
        const int* __restrict__ obsg, const float* __restrict__ initk,
        float* __restrict__ out) {
    extern __shared__ float lds[];
    float* Btl  = lds;                       // ECOLS*SPAD
    float* bufA = Btl + ECOLS * SPAD;        // SPAD
    float* bufB = bufA + SPAD;               // SPAD
    float* red  = bufB + SPAD;               // 64
    int*   obsl = (int*)(red + 64);          // TLEN

    const int tid = threadIdx.x;
    const int b   = blockIdx.x;
    const int j   = tid;

#ifdef HAVE_TDM
    // wave 0 kicks a single TDM DMA of the 249KB emission matrix into LDS...
    if (tid < 32) tdm_load_bt(Btg, Btl);
#else
    for (int i = tid; i < ECOLS * SPAD; i += NTHR) Btl[i] = Btg[i];
#endif
    // ...overlapped with all waves staging this sequence's obs stream
    const int* osrc = obsg + (long)b * TLEN;
    for (int i = tid; i < TLEN; i += NTHR) obsl[i] = osrc[i];

    // per-thread transition taps (constant for the whole recurrence)
    float c0 = 0, c1 = 0, c2 = 0, c3 = 0;
    int jm1 = 0, jm2 = 0, jm3 = 0;
    if (j < SDIM) {
        c0 = bandsg[0 * SPAD + j]; c1 = bandsg[1 * SPAD + j];
        c2 = bandsg[2 * SPAD + j]; c3 = bandsg[3 * SPAD + j];
        jm1 = (j + SDIM - 1) % SDIM;
        jm2 = (j + SDIM - 2) % SDIM;
        jm3 = (j + SDIM - 3) % SDIM;
    }
#ifdef HAVE_TDM
    if (tid < 32) tdm_wait0();               // drain TENSORcnt on issuing wave
#endif
    __syncthreads();                         // Bt + obs now visible to all waves

    float* cur = bufA;
    float* nxt = bufB;

    // t = 0: alpha = Bt[obs0] * I, I = softmax(init_kernel) on states 0..3
    float av = 0.0f;
    {
        int o0 = obsl[0];
        if (j < 4) {
            float g0 = initk[0], g1 = initk[1], g2 = initk[2], g3 = initk[3];
            float m = fmaxf(fmaxf(g0, g1), fmaxf(g2, g3));
            float s = expf(g0 - m) + expf(g1 - m) + expf(g2 - m) + expf(g3 - m);
            float Ij = expf(initk[j] - m) / s;
            av = Btl[o0 * SPAD + j] * Ij;
        }
        if (j < SDIM) cur[j] = av;
    }
    __syncthreads();
    float ll;
    {
        float z = block_sum(av, red);
        ll = logf(z);
        if (j < SDIM) cur[j] = av / z;
        av /= z;
    }
    __syncthreads();

    // main recurrence
    for (int t = 1; t < TLEN; ++t) {
        int o = obsl[t];
        av = 0.0f;
        if (j < SDIM) {
            float e = Btl[o * SPAD + j];
            av = e * (c0 * cur[j] + c1 * cur[jm1] + c2 * cur[jm2] + c3 * cur[jm3]);
            nxt[j] = av;
        }
        __syncthreads();
        { float* tmp = cur; cur = nxt; nxt = tmp; }
        if ((t & 7) == 7 && t != TLEN - 1) {   // periodic rescale (log-equivalent)
            float z = block_sum(av, red);
            ll += logf(z);
            float inv = 1.0f / z;
            av *= inv;
            if (j < SDIM) cur[j] = av;
            __syncthreads();
        }
    }
    float zf = block_sum(av, red);
    ll += logf(zf);
    if (tid == 0) out[b] = ll;
}

// ---------------------------------------------------------------------
extern "C" void kernel_launch(void* const* d_in, const int* in_sizes, int n_in,
                              void* d_out, int out_size, void* d_ws, size_t ws_size,
                              hipStream_t stream) {
    const float* x  = (const float*)d_in[0];   // [B,T,101] one-hot f32
    const float* ik = (const float*)d_in[1];   // [4] init kernel
    const float* tk = (const float*)d_in[2];   // [S*3] transition kernel
    const float* ek = (const float*)d_in[3];   // [611*24*4] emission kernel
    float* out = (float*)d_out;                // [B] loglik

    // workspace layout
    char*  ws    = (char*)d_ws;
    float* bands = (float*)ws;                                            // 4*SPAD f32
    float* Btg   = (float*)(ws + (size_t)4 * SPAD * sizeof(float));       // ECOLS*SPAD f32
    int*   obsg  = (int*)(ws + (size_t)(4 * SPAD + ECOLS * SPAD) * sizeof(float)); // B*T i32

    const int nbt = ECOLS * SPAD;
    zero_f32<<<(nbt + 255) / 256, 256, 0, stream>>>(Btg, nbt);
    build_bands<<<(SDIM + 255) / 256, 256, 0, stream>>>(tk, bands);
    build_emis<<<(NSC + 1 + 255) / 256, 256, 0, stream>>>(ek, Btg);

    const int tiles = (NBATCH * TLEN) / 16;    // 16384 tiles, 8 waves/block
    obs_extract_wmma<<<tiles / 8, 256, 0, stream>>>(x, obsg);

    const size_t shmem = (size_t)(ECOLS * SPAD + 2 * SPAD + 64) * sizeof(float)
                       + (size_t)TLEN * sizeof(int);   // ~287 KB (<320 KB/WGP)
    hipFuncSetAttribute(reinterpret_cast<const void*>(&forward_kernel),
                        hipFuncAttributeMaxDynamicSharedMemorySize, (int)shmem);
    forward_kernel<<<NBATCH, NTHR, shmem, stream>>>(Btg, bands, obsg, ik, out);
}
